// NeighborAttention_32263794328106
// MI455X (gfx1250) — matrix operand order repaired and checked by reading, not compiled, not run on hardware
//
#include <hip/hip_runtime.h>
#include <hip/hip_bf16.h>
#include <stdint.h>

typedef __bf16 bf16;
typedef __attribute__((ext_vector_type(16))) __bf16 v16bf;
typedef __attribute__((ext_vector_type(8)))  float  v8f;

// Problem dimensions (fixed by the reference)
#define BB   4
#define NN   2000
#define KKN  30     // neighbors
#define DD   128
#define DIN  256
#define HH   4
#define HD   32
#define SITES (BB * NN)   // 8000

// ---------------------------------------------------------------------------
// f32 -> bf16 with round-to-nearest-even (bit twiddle; no __bf16 arithmetic)
// ---------------------------------------------------------------------------
__device__ __forceinline__ bf16 f2bf(float f) {
  union { float f; uint32_t u; } a; a.f = f;
  uint32_t u = a.u + 0x7FFFu + ((a.u >> 16) & 1u);
  union { uint16_t s; bf16 b; } r; r.s = (uint16_t)(u >> 16);
  return r.b;
}

// A-fragment K index map (contiguous, matches B fills):
// lanes 0-15 (g=0): elements 0..15 -> K 0..15
// lanes 16-31 (g=1): elements 0..15 -> K 16..31
__device__ __forceinline__ int kmapA(int e, int g) {
  return 16 * g + e;
}

// ---------------------------------------------------------------------------
// Generic tiled GEMM: C[M x N] = A[M x Kd] * B, B[k][n] taken from f32 W as
//   W_TRANS ? W[rowOff + n][k]  (C = A * W^T)
//           : W[rowOff + k][n]  (C = A * W)
// BM = 64 rows/block, BN = NTILES*16 cols/block, 256 threads = 8 waves (4x2).
// blockIdx.z selects a "head" via the *Z offsets.
// ---------------------------------------------------------------------------
template <int NTILES, bool A_F32, bool W_TRANS, bool OUT_F32>
__global__ __launch_bounds__(256)
void wmma_gemm(const void* __restrict__ Av, int aStride, int aColOff, int aColZ,
               const float* __restrict__ W, int wStride, int wRowOff, int wRowZ,
               void* __restrict__ Cv, int cStride, int cColOff, int cColZ,
               int Kd) {
  constexpr int BN = NTILES * 16;
  constexpr int FR = NTILES / 2;   // C fragments per wave

  __shared__ bf16 As[64][40];      // 64 x 32 (+pad)
  __shared__ bf16 Bs[32][BN + 8];  // 32 x BN (+pad)

  const int tid  = threadIdx.x;
  const int lane = tid & 31;
  const int wid  = tid >> 5;
  const int wm   = wid & 3;        // wave row (4 x 16 = 64 M)
  const int wn   = wid >> 2;       // wave col (2 halves of BN)
  const int z    = blockIdx.z;
  const int mBase = blockIdx.x * 64;
  const int nBase = blockIdx.y * BN;

  const float* Af = (const float*)Av;
  const bf16*  Ab = (const bf16*)Av;

  v8f acc[FR];
#pragma unroll
  for (int f = 0; f < FR; ++f)
#pragma unroll
    for (int i = 0; i < 8; ++i) acc[f][i] = 0.0f;

  const int aOff  = aColOff + z * aColZ;
  const int wOffR = wRowOff + z * wRowZ;

  for (int k0 = 0; k0 < Kd; k0 += 32) {
    // ---- stage A tile (64x32), converting f32->bf16 if needed ----
#pragma unroll
    for (int i = 0; i < 8; ++i) {
      int idx = tid + i * 256;
      int kk = idx & 31, mm = idx >> 5;
      size_t gi = (size_t)(mBase + mm) * aStride + aOff + k0 + kk;
      As[mm][kk] = A_F32 ? f2bf(Af[gi]) : Ab[gi];
    }
    // ---- stage B tile (32 x BN) from f32 weights ----
#pragma unroll
    for (int i = 0; i < (32 * BN) / 256; ++i) {
      int idx = tid + i * 256;
      int kk, nn;
      if (W_TRANS) { kk = idx & 31; nn = idx >> 5; }
      else         { nn = idx % BN; kk = idx / BN; }
      size_t gi = W_TRANS
          ? (size_t)(wOffR + nBase + nn) * wStride + (size_t)(k0 + kk)
          : (size_t)(wOffR + k0 + kk) * wStride + (size_t)(nBase + nn);
      Bs[kk][nn] = f2bf(W[gi]);
    }
    __syncthreads();

    // ---- WMMA: each wave does one A-frag x FR B-frags ----
    const int g  = lane >> 4;
    const int ml = wm * 16 + (lane & 15);
    v16bf af;
#pragma unroll
    for (int e = 0; e < 16; ++e) af[e] = As[ml][kmapA(e, g)];
#pragma unroll
    for (int f = 0; f < FR; ++f) {
      const int nt = wn * FR + f;
      v16bf bfr;
#pragma unroll
      for (int e = 0; e < 16; ++e) bfr[e] = Bs[16 * g + e][nt * 16 + (lane & 15)];
      acc[f] = __builtin_amdgcn_wmma_f32_16x16x32_bf16(
          false, af, false, bfr, (short)0, acc[f], false, false);
    }
    __syncthreads();
  }

  // ---- store C fragments: element (M = r + 8*(lane>=16), N = lane&15) ----
  float* Cf = (float*)Cv;
  bf16*  Cb = (bf16*)Cv;
  const int g = lane >> 4;
#pragma unroll
  for (int f = 0; f < FR; ++f) {
    const int nt  = wn * FR + f;
    const int col = cColOff + z * cColZ + nBase + nt * 16 + (lane & 15);
#pragma unroll
    for (int r = 0; r < 8; ++r) {
      const int row = mBase + wm * 16 + r + 8 * g;
      size_t ci = (size_t)row * cStride + col;
      if (OUT_F32) Cf[ci] = acc[f][r];
      else         Cb[ci] = f2bf(acc[f][r]);
    }
  }
}

// ---------------------------------------------------------------------------
// Per-site attention core (one block per (b,n) site, 256 threads = 8 waves):
//   logits[k,h] = (h_E[s] @ qk[s]^T)/sqrt(HD)  via WMMA, K-chunks over waves
//   masked softmax (f32)
//   hbar[h,:]   = w @ h_E[s]                    via WMMA, col tiles over waves
// ---------------------------------------------------------------------------
__global__ __launch_bounds__(256)
void attn_core(const float* __restrict__ hE, const int* __restrict__ mask,
               const bf16* __restrict__ qk, bf16* __restrict__ hbar) {
  __shared__ bf16  Es[32][DIN + 8];   // neighbors (zero-padded 30->32) x 256
  __shared__ bf16  Qs[16][DIN + 8];   // heads (zero-padded 4->16) x 256
  __shared__ float lp[8][32][4];      // per-wave partial logits
  __shared__ float ml[32][4];         // masked logits / exp scratch
  __shared__ bf16  wS[16][40];        // softmax weights (A-frag source)

  const int s    = blockIdx.x;
  const int tid  = threadIdx.x;
  const int lane = tid & 31;
  const int wid  = tid >> 5;

  // stage h_E site -> bf16 (pad rows 30,31 with zero)
  for (int idx = tid; idx < 32 * DIN; idx += 256) {
    int r = idx >> 8, c = idx & 255;
    float v = (r < KKN) ? hE[((size_t)s * KKN + r) * DIN + c] : 0.0f;
    Es[r][c] = f2bf(v);
  }
  // stage qk site (pad rows 4..15 with zero)
  for (int idx = tid; idx < 16 * DIN; idx += 256) {
    int r = idx >> 8, c = idx & 255;
    Qs[r][c] = (r < HH) ? qk[(size_t)s * (HH * DIN) + r * DIN + c] : f2bf(0.0f);
  }
  for (int idx = tid; idx < 16 * 32; idx += 256)
    wS[idx >> 5][idx & 31] = f2bf(0.0f);
  __syncthreads();

  // ---- logits^T (32 neighbors x 16 headslots): wave w owns K-chunk w*32 ----
  {
    const int c0 = wid * 32;
    const int g  = lane >> 4;
    const int nh = lane & 15;
#pragma unroll
    for (int mt = 0; mt < 2; ++mt) {
      const int m = mt * 16 + (lane & 15);
      v16bf af;
#pragma unroll
      for (int e = 0; e < 16; ++e) af[e] = Es[m][c0 + kmapA(e, g)];
      v16bf bfr;
#pragma unroll
      for (int e = 0; e < 16; ++e) bfr[e] = Qs[nh][c0 + 16 * g + e];
      v8f c;
#pragma unroll
      for (int i = 0; i < 8; ++i) c[i] = 0.0f;
      c = __builtin_amdgcn_wmma_f32_16x16x32_bf16(
          false, af, false, bfr, (short)0, c, false, false);
      if (nh < 4) {
#pragma unroll
        for (int r = 0; r < 8; ++r)
          lp[wid][mt * 16 + r + 8 * g][nh] = c[r];
      }
    }
  }
  __syncthreads();

  // ---- reduce K-chunk partials, scale, apply mask ----
  if (tid < 128) {
    int k = tid >> 2, h = tid & 3;
    float v = 0.0f;
#pragma unroll
    for (int w = 0; w < 8; ++w) v += lp[w][k][h];
    v *= 0.17677669529663687f;  // 1/sqrt(32)
    const float NEG = -3.402823466e38f;
    if (k < KKN) v = (mask[(size_t)s * KKN + k] != 0) ? v : NEG;
    else         v = NEG;
    ml[k][h] = v;
  }
  __syncthreads();

  // ---- masked softmax per head; weights (x mask) -> wS (bf16, zero-padded) --
  if (tid < HH) {
    const int h = tid;
    float mx = -3.402823466e38f;
    for (int k = 0; k < KKN; ++k) mx = fmaxf(mx, ml[k][h]);
    float sum = 0.0f;
    for (int k = 0; k < KKN; ++k) {
      float e = __expf(ml[k][h] - mx);
      ml[k][h] = e;
      sum += e;
    }
    float inv = 1.0f / sum;
    for (int k = 0; k < KKN; ++k) {
      float w = (mask[(size_t)s * KKN + k] != 0) ? ml[k][h] * inv : 0.0f;
      wS[h][k] = f2bf(w);
    }
  }
  __syncthreads();

  // ---- hbar (16x256) = wS (16x32) x Es (32x256): 2 col tiles per wave ----
  {
    const int g = lane >> 4;
    v16bf af;
#pragma unroll
    for (int e = 0; e < 16; ++e) af[e] = wS[lane & 15][kmapA(e, g)];
#pragma unroll
    for (int j = 0; j < 2; ++j) {
      const int nt = wid * 2 + j;
      v16bf bfr;
#pragma unroll
      for (int e = 0; e < 16; ++e) bfr[e] = Es[16 * g + e][nt * 16 + (lane & 15)];
      v8f c;
#pragma unroll
      for (int i = 0; i < 8; ++i) c[i] = 0.0f;
      c = __builtin_amdgcn_wmma_f32_16x16x32_bf16(
          false, af, false, bfr, (short)0, c, false, false);
      if (lane < 16) {   // rows M = r (r<4 are real heads), lanes 0..15 = cols
#pragma unroll
        for (int r = 0; r < 4; ++r)
          hbar[(size_t)s * (HH * DIN) + r * DIN + nt * 16 + lane] = f2bf(c[r]);
      }
    }
  }
}

// ---------------------------------------------------------------------------
// Launch: 5 kernels, ~37 MB workspace of bf16 intermediates.
// ---------------------------------------------------------------------------
extern "C" void kernel_launch(void* const* d_in, const int* in_sizes, int n_in,
                              void* d_out, int out_size, void* d_ws, size_t ws_size,
                              hipStream_t stream) {
  (void)in_sizes; (void)n_in; (void)out_size; (void)ws_size;
  const float* h_V = (const float*)d_in[0];
  const float* h_E = (const float*)d_in[1];
  const int*   msk = (const int*)d_in[2];
  const float* W_Q = (const float*)d_in[3];
  const float* W_K = (const float*)d_in[4];
  const float* W_V = (const float*)d_in[5];
  const float* W_O = (const float*)d_in[6];
  float* out = (float*)d_out;

  char* ws = (char*)d_ws;
  bf16* Qb   = (bf16*)(ws);                               //  2,048,000 B
  bf16* qkb  = (bf16*)(ws + 2048000);                     // 16,384,000 B
  bf16* hbar = (bf16*)(ws + 2048000 + 16384000);          // 16,384,000 B
  bf16* hupd = (bf16*)(ws + 2048000 + 2 * 16384000);      //  2,048,000 B

  dim3 blk(256);

  // K1: Q = h_V @ W_Q^T          (M=8000, N=128, K=128) -> bf16
  wmma_gemm<8, true, true, false><<<dim3(SITES / 64, 1, 1), blk, 0, stream>>>(
      h_V, DD, 0, 0, W_Q, DD, 0, 0, Qb, DD, 0, 0, DD);

  // K2: qk[s,h,:] = Q[s,hblock] @ W_K[hblock,:]   (per head, K=32, N=256)
  wmma_gemm<8, false, false, false><<<dim3(SITES / 64, DIN / 128, HH), blk, 0, stream>>>(
      Qb, DD, 0, HD, W_K, DIN, 0, HD, qkb, HH * DIN, 0, DIN, HD);

  // K3: per-site logits -> masked softmax -> hbar
  attn_core<<<dim3(SITES), blk, 0, stream>>>(h_E, msk, qkb, hbar);

  // K4: hupd[s,hblock] = hbar[s,h,:] @ W_V[hblock,:]^T  (per head, K=256, N=32)
  wmma_gemm<2, false, true, false><<<dim3(SITES / 64, 1, HH), blk, 0, stream>>>(
      hbar, HH * DIN, 0, DIN, W_V, DIN, 0, HD, hupd, DD, 0, HD, DIN);

  // K5: out = hupd @ W_O^T       (M=8000, N=128, K=128) -> f32
  wmma_gemm<8, false, true, true><<<dim3(SITES / 64, 1, 1), blk, 0, stream>>>(
      hupd, DD, 0, 0, W_O, DD, 0, 0, out, DD, 0, 0, DD);
}
